// GraphVectorQuantizer_63144609185895
// MI455X (gfx1250) — matrix operand order repaired and checked by reading, not compiled, not run on hardware
//
#include <hip/hip_runtime.h>
#include <hip/hip_bf16.h>

#define N_NODES       65536
#define N_EDGES       1048576
#define K_CODES       8192
#define DIM           512
#define ROWS_PER_WAVE 16
#define WAVES_PER_WG  4
#define ROWS_PER_WG   (ROWS_PER_WAVE * WAVES_PER_WG)

typedef __attribute__((ext_vector_type(16))) __bf16          v16bf;
typedef __attribute__((ext_vector_type(16))) unsigned short  v16u;
typedef __attribute__((ext_vector_type(8)))  float           v8f;

// ---------------------------------------------------------------------------
// round-to-nearest-even f32 -> bf16 bits
__device__ __forceinline__ unsigned short f32_to_bf16_rne(float f) {
    unsigned int u = __float_as_uint(f);
    u += 0x7FFFu + ((u >> 16) & 1u);
    return (unsigned short)(u >> 16);
}

// ---------------------------------------------------------------------------
// Kernel 1: emb f32 -> bf16 (workspace), per-code squared norm, zero loss slot
__global__ void vq_prep_kernel(const float* __restrict__ emb,
                               unsigned short* __restrict__ emb_bf16,
                               float* __restrict__ emb_norm,
                               float* __restrict__ loss_slot) {
    const int row = blockIdx.x;                      // one block per code
    const float* er = emb + (size_t)row * DIM;
    unsigned short* br = emb_bf16 + (size_t)row * DIM;

    float s = 0.f;
    for (int c = threadIdx.x; c < DIM; c += blockDim.x) {
        float v = er[c];
        s = fmaf(v, v, s);
        br[c] = f32_to_bf16_rne(v);
    }
    __shared__ float red[256];
    red[threadIdx.x] = s;
    __syncthreads();
    for (int st = 128; st >= 1; st >>= 1) {
        if ((int)threadIdx.x < st) red[threadIdx.x] += red[threadIdx.x + st];
        __syncthreads();
    }
    if (threadIdx.x == 0) {
        emb_norm[row] = red[0];
        if (row == 0) *loss_slot = 0.f;
    }
}

// ---------------------------------------------------------------------------
// Kernel 2: fused distance-GEMM (bf16 WMMA, f32 accum) + argmin + gather + loss
__global__ void __launch_bounds__(WAVES_PER_WG * 32)
vq_assign_kernel(const float* __restrict__ x,
                 const float* __restrict__ emb,             // f32 codebook (gather)
                 const unsigned short* __restrict__ emb16,  // bf16 codebook (GEMM)
                 const float* __restrict__ emb_norm,        // ||e||^2
                 float* __restrict__ zq_out,                // [N, D]
                 float* __restrict__ idx_out_f,             // [N] (as float)
                 float* __restrict__ loss_slot) {
    const int lane   = threadIdx.x & 31;
    const int wave   = threadIdx.x >> 5;
    const int half   = lane >> 4;          // 0: lanes 0-15, 1: lanes 16-31
    const int mLane  = lane & 15;          // row within 16-row tile (A side)
    const int row0   = blockIdx.x * ROWS_PER_WG + wave * ROWS_PER_WAVE;

    // --- Load this wave's 16 x-rows as persistent bf16 A fragments ----------
    // A 16x32 layout: lane<16 holds M=lane, elems 0..15 = K k0..k0+15;
    // lane>=16 holds K k0+16..k0+31 (contiguous, matching B).
    v16bf afrag[DIM / 32];
    {
        const float* xrow = x + (size_t)(row0 + mLane) * DIM;
#pragma unroll
        for (int kt = 0; kt < DIM / 32; ++kt) {
            const int k0 = kt * 32;
            const int kA = k0 + half * 16;       // elems 0..7
            const int kB = k0 + half * 16 + 8;   // elems 8..15
            v16u t;
#pragma unroll
            for (int e = 0; e < 8; ++e) {
                t[e]     = f32_to_bf16_rne(xrow[kA + e]);
                t[e + 8] = f32_to_bf16_rne(xrow[kB + e]);
            }
            afrag[kt] = __builtin_bit_cast(v16bf, t);
        }
    }

    // --- Shared staging for one 16-code x 512-dim bf16 codebook tile --------
    __shared__ __align__(32) unsigned short btile[16 * DIM];   // 16 KB

    float bestv[8];
    int   besti[8];
#pragma unroll
    for (int r = 0; r < 8; ++r) { bestv[r] = 3.4e38f; besti[r] = 0; }

    for (int n0 = 0; n0 < K_CODES; n0 += 16) {
        // Cooperative stage of B tile (16 codes are contiguous in emb16)
        __syncthreads();
        {
            const uint4* src = (const uint4*)(emb16 + (size_t)n0 * DIM);
            uint4* dst = (uint4*)btile;
            const int nvec = (16 * DIM * 2) / 16;   // 1024 uint4
            for (int i = threadIdx.x; i < nvec; i += blockDim.x) dst[i] = src[i];
            // prefetch next tile into cache hierarchy
            if (n0 + 16 < K_CODES)
                __builtin_prefetch(emb16 + (size_t)(n0 + 16) * DIM + threadIdx.x * 64, 0, 0);
        }
        __syncthreads();

        // 16x16 distance tile: acc = x . e^T  (f32 accumulate)
        v8f acc = {};
#pragma unroll
        for (int kt = 0; kt < DIM / 32; ++kt) {
            // B 32x16 layout: lane n=lane&15; lanes 0-15 hold K k0..k0+15,
            // lanes 16-31 hold K k0+16..k0+31 (16 contiguous bf16 per lane).
            const v16bf* bp =
                (const v16bf*)(btile + (size_t)mLane * DIM + kt * 32 + half * 16);
            v16bf b = *bp;
            acc = __builtin_amdgcn_wmma_f32_16x16x32_bf16(
                false, afrag[kt], false, b, (short)0, acc, false, false);
        }

        // d = ||e||^2 - 2 x.e   (||x||^2 is row-constant: argmin-invariant)
        const float enorm = emb_norm[n0 + mLane];
        const int   ci    = n0 + mLane;
#pragma unroll
        for (int r = 0; r < 8; ++r) {
            const float dval = fmaf(-2.f, acc[r], enorm);
            if (dval < bestv[r] || (dval == bestv[r] && ci < besti[r])) {
                bestv[r] = dval;
                besti[r] = ci;
            }
        }
    }

    // --- Reduce (min,idx) across the 16 lanes of each half ------------------
    // C layout: lanes 0-15 VGPR r <-> row r ; lanes 16-31 VGPR r <-> row r+8.
#pragma unroll
    for (int r = 0; r < 8; ++r) {
        float v = bestv[r];
        int   i = besti[r];
#pragma unroll
        for (int off = 8; off >= 1; off >>= 1) {   // xor 8,4,2,1 stays in-half
            const float ov = __shfl_xor(v, off, 32);
            const int   oi = __shfl_xor(i, off, 32);
            if (ov < v || (ov == v && oi < i)) { v = ov; i = oi; }
        }
        bestv[r] = v;
        besti[r] = i;
    }

    // Broadcast per-row argmin to all lanes of the wave
    int idxAll[16];
#pragma unroll
    for (int r = 0; r < 8; ++r) {
        idxAll[r]     = __shfl(besti[r], 0, 32);    // rows 0..7
        idxAll[r + 8] = __shfl(besti[r], 16, 32);   // rows 8..15
    }

    if (lane == 0) {
#pragma unroll
        for (int r = 0; r < 16; ++r)
            idx_out_f[row0 + r] = (float)idxAll[r];
    }

    // --- Gather z_q = emb[idx] (f32), accumulate loss -----------------------
    float lsum = 0.f;
#pragma unroll 1
    for (int r = 0; r < 16; ++r) {
        const float* erow = emb + (size_t)idxAll[r] * DIM;
        const float* xrow = x + (size_t)(row0 + r) * DIM;
        float* orow = zq_out + (size_t)(row0 + r) * DIM;
        for (int c = lane; c < DIM; c += 32) {
            const float e  = erow[c];
            const float xv = xrow[c];
            orow[c] = e;                      // straight-through: z_q == emb[idx]
            const float df = e - xv;
            lsum = fmaf(df, df, lsum);
        }
    }
#pragma unroll
    for (int off = 16; off >= 1; off >>= 1) lsum += __shfl_xor(lsum, off, 32);
    if (lane == 0) {
        // loss = mean(d^2) + 0.25*mean(d^2) = 1.25 * mean(d^2)
        atomicAdd(loss_slot, lsum * (1.25f / ((float)N_NODES * (float)DIM)));
    }
}

// ---------------------------------------------------------------------------
// Kernel 3: edge_index int64 -> f32 pass-through
__global__ void vq_edge_kernel(const long long* __restrict__ e,
                               float* __restrict__ out, int n) {
    const int i = blockIdx.x * blockDim.x + threadIdx.x;
    if (i < n) out[i] = (float)e[i];
}

// ---------------------------------------------------------------------------
extern "C" void kernel_launch(void* const* d_in, const int* in_sizes, int n_in,
                              void* d_out, int out_size, void* d_ws, size_t ws_size,
                              hipStream_t stream) {
    (void)in_sizes; (void)n_in; (void)out_size; (void)ws_size;

    const float*     x    = (const float*)d_in[0];
    const long long* edge = (const long long*)d_in[1];
    const float*     emb  = (const float*)d_in[2];

    // Output: [z_q (N*D)] [loss (1)] [indices (N)] [edge (2*E)]  (all f32)
    float* out       = (float*)d_out;
    float* zq        = out;
    float* loss_slot = out + (size_t)N_NODES * DIM;
    float* idx_out   = loss_slot + 1;
    float* edge_out  = idx_out + N_NODES;

    // Workspace: bf16 codebook (8 MB) + code norms (32 KB)
    unsigned short* emb16    = (unsigned short*)d_ws;
    float*          emb_norm = (float*)((char*)d_ws + (size_t)K_CODES * DIM * 2);

    vq_prep_kernel<<<K_CODES, 256, 0, stream>>>(emb, emb16, emb_norm, loss_slot);

    vq_assign_kernel<<<N_NODES / ROWS_PER_WG, WAVES_PER_WG * 32, 0, stream>>>(
        x, emb, emb16, emb_norm, zq, idx_out, loss_slot);

    const int nEdge = 2 * N_EDGES;
    vq_edge_kernel<<<(nEdge + 255) / 256, 256, 0, stream>>>(edge, edge_out, nEdge);
}
